// Attention_69655779606628
// MI455X (gfx1250) — matrix operand order repaired and checked by reading, not compiled, not run on hardware
//
#include <hip/hip_runtime.h>
#include <hip/hip_bf16.h>

typedef __attribute__((ext_vector_type(16))) __bf16 v16bf;
typedef __attribute__((ext_vector_type(8)))  float  v8f;
typedef __attribute__((ext_vector_type(4)))  int    v4i_;
typedef __attribute__((address_space(1))) v4i_ gv4i;
typedef __attribute__((address_space(3))) v4i_ lv4i;

union Frag16 {
  v16bf v;
  uint4 q[2];
};

__device__ __forceinline__ v8f wmma_bf16(v16bf a, v16bf b, v8f c) {
  // (neg_a, A, neg_b, B, c_mod, C, reuse_a, reuse_b)
  return __builtin_amdgcn_wmma_f32_16x16x32_bf16(false, a, false, b, (short)0, c,
                                                 false, false);
}

#if defined(__has_builtin)
#if __has_builtin(__builtin_amdgcn_global_load_async_to_lds_b128)
#define HAVE_ASYNC_LDS 1
#endif
#endif

#ifdef HAVE_ASYNC_LDS
__device__ __forceinline__ void wait_asynccnt0() {
#if __has_builtin(__builtin_amdgcn_s_wait_asynccnt)
  __builtin_amdgcn_s_wait_asynccnt(0);
#else
  asm volatile("s_wait_asynccnt 0" ::: "memory");
#endif
}
#endif

// 16 consecutive bf16 (32B), 16B-aligned -> B fragment
__device__ __forceinline__ v16bf load16(const __bf16* p) {
  Frag16 f;
  f.q[0] = *(const uint4*)(p);
  f.q[1] = *(const uint4*)(p + 8);
  return f.v;
}

// A fragment: elements j=0..7 at p, j=8..15 at p+16 (two groups of 8 consecutive K)
__device__ __forceinline__ v16bf load8x2(const __bf16* p) {
  Frag16 f;
  f.q[0] = *(const uint4*)(p);
  f.q[1] = *(const uint4*)(p + 16);
  return f.v;
}

#define B_  8
#define C_  256
#define N_  4096

// ---------------------------------------------------------------------------
// Pre-convert the four 256x256 f32 weight matrices to bf16.
// ---------------------------------------------------------------------------
__global__ __launch_bounds__(256) void sagan_wcvt(
    const float* __restrict__ wq, const float* __restrict__ wk,
    const float* __restrict__ wv, const float* __restrict__ wo,
    __bf16* __restrict__ Wqb, __bf16* __restrict__ Wkb,
    __bf16* __restrict__ Wvb, __bf16* __restrict__ Wob) {
  int i = blockIdx.x * 256 + threadIdx.x;  // float4 index, 16384 total
  union { uint2 u; __bf16 h[4]; } p;
  float4 a;
  a = ((const float4*)wq)[i];
  p.h[0] = (__bf16)a.x; p.h[1] = (__bf16)a.y; p.h[2] = (__bf16)a.z; p.h[3] = (__bf16)a.w;
  ((uint2*)Wqb)[i] = p.u;
  a = ((const float4*)wk)[i];
  p.h[0] = (__bf16)a.x; p.h[1] = (__bf16)a.y; p.h[2] = (__bf16)a.z; p.h[3] = (__bf16)a.w;
  ((uint2*)Wkb)[i] = p.u;
  a = ((const float4*)wv)[i];
  p.h[0] = (__bf16)a.x; p.h[1] = (__bf16)a.y; p.h[2] = (__bf16)a.z; p.h[3] = (__bf16)a.w;
  ((uint2*)Wvb)[i] = p.u;
  a = ((const float4*)wo)[i];
  p.h[0] = (__bf16)a.x; p.h[1] = (__bf16)a.y; p.h[2] = (__bf16)a.z; p.h[3] = (__bf16)a.w;
  ((uint2*)Wob)[i] = p.u;
}

// ---------------------------------------------------------------------------
// Per-channel BN statistics (mean, rstd) over [b, n]
// ---------------------------------------------------------------------------
__global__ __launch_bounds__(256) void sagan_bn_stats(const float* __restrict__ x,
                                                      float* __restrict__ mean,
                                                      float* __restrict__ rstd) {
  const int ch = blockIdx.x;
  const int tid = threadIdx.x;
  const int lane = tid & 31, wave = tid >> 5;
  float s1 = 0.f, s2 = 0.f;
  for (int bi = 0; bi < B_; ++bi) {
    const float4* p = (const float4*)(x + ((size_t)bi * C_ + ch) * N_);
    for (int t = tid; t < N_ / 4; t += 256) {
      float4 v = p[t];
      s1 += v.x + v.y + v.z + v.w;
      s2 += v.x * v.x + v.y * v.y + v.z * v.z + v.w * v.w;
    }
  }
  for (int m = 1; m <= 16; m <<= 1) {
    s1 += __shfl_xor(s1, m, 32);
    s2 += __shfl_xor(s2, m, 32);
  }
  __shared__ float a1[8], a2[8];
  if (lane == 0) { a1[wave] = s1; a2[wave] = s2; }
  __syncthreads();
  if (tid == 0) {
    float t1 = 0.f, t2 = 0.f;
    for (int i = 0; i < 8; ++i) { t1 += a1[i]; t2 += a2[i]; }
    const float inv = 1.0f / (float)(B_ * N_);
    float m = t1 * inv;
    mean[ch] = m;
    rstd[ch] = rsqrtf(t2 * inv - m * m + 1e-5f);
  }
}

// ---------------------------------------------------------------------------
// One fused BN+ReLU+GEMM branch. HT tile: [token][channel] bf16, stride 264.
// ---------------------------------------------------------------------------
__device__ __forceinline__ void proj_branch(
    const float* __restrict__ x, const __bf16* __restrict__ Wb,
    const float* __restrict__ g, const float* __restrict__ bt,
    const float* __restrict__ mean, const float* __restrict__ rstd,
    __bf16* __restrict__ Ob, __bf16 (*HT)[264],
    int bi, int t0, int tid, int lane, int wave, int half, int col) {
  for (int idx = tid; idx < 64 * 256 / 4; idx += 256) {
    int tok4 = idx & 15;
    int ch = idx >> 4;
    float4 xv = *(const float4*)&x[((size_t)bi * C_ + ch) * N_ + t0 + tok4 * 4];
    float gg = g[ch], bb = bt[ch], mm = mean[ch], rr = rstd[ch];
    HT[tok4 * 4 + 0][ch] = (__bf16)fmaxf((xv.x - mm) * rr * gg + bb, 0.f);
    HT[tok4 * 4 + 1][ch] = (__bf16)fmaxf((xv.y - mm) * rr * gg + bb, 0.f);
    HT[tok4 * 4 + 2][ch] = (__bf16)fmaxf((xv.z - mm) * rr * gg + bb, 0.f);
    HT[tok4 * 4 + 3][ch] = (__bf16)fmaxf((xv.w - mm) * rr * gg + bb, 0.f);
  }
  __syncthreads();

  for (int oi = wave * 2; oi < wave * 2 + 2; ++oi) {
    v16bf afr[8];
    const __bf16* Wrow = Wb + (size_t)(oi * 16 + col) * C_;
    for (int kk = 0; kk < 8; ++kk) afr[kk] = load16(Wrow + kk * 32 + half * 16);
    for (int tt = 0; tt < 4; ++tt) {
      v8f acc;
      for (int i = 0; i < 8; ++i) acc[i] = 0.f;
      for (int kk = 0; kk < 8; ++kk) {
        v16bf bf = load16(&HT[tt * 16 + col][kk * 32 + half * 16]);
        acc = wmma_bf16(afr[kk], bf, acc);
      }
      int tok = t0 + tt * 16 + col;
      union { uint4 u; __bf16 h[8]; } pk;
      for (int r = 0; r < 8; ++r) pk.h[r] = (__bf16)acc[r];
      *(uint4*)&Ob[((size_t)bi * N_ + tok) * C_ + oi * 16 + 8 * half] = pk.u;
    }
  }
  __syncthreads();
}

__global__ __launch_bounds__(256) void sagan_qkv_proj(
    const float* __restrict__ x,
    const __bf16* __restrict__ Wqb, const __bf16* __restrict__ Wkb,
    const __bf16* __restrict__ Wvb,
    const float* __restrict__ gq, const float* __restrict__ bq,
    const float* __restrict__ gk, const float* __restrict__ bk,
    const float* __restrict__ gv, const float* __restrict__ bv,
    const float* __restrict__ mean, const float* __restrict__ rstd,
    __bf16* __restrict__ Qb, __bf16* __restrict__ Kb, __bf16* __restrict__ Vb) {
  const int bi = blockIdx.x;
  const int t0 = blockIdx.y * 64;
  const int tid = threadIdx.x;
  const int lane = tid & 31, wave = tid >> 5;
  const int half = (lane >> 4) & 1;
  const int col = lane & 15;

  __shared__ __bf16 HT[64][264];

  proj_branch(x, Wqb, gq, bq, mean, rstd, Qb, HT, bi, t0, tid, lane, wave, half, col);
  proj_branch(x, Wkb, gk, bk, mean, rstd, Kb, HT, bi, t0, tid, lane, wave, half, col);
  proj_branch(x, Wvb, gv, bv, mean, rstd, Vb, HT, bi, t0, tid, lane, wave, half, col);
}

// ---------------------------------------------------------------------------
// Flash attention with producer/consumer wave pairs.
// 512 threads = 16 waves = 8 pairs; a pair owns 16 query rows.
// Even wave: S = Q*K^T (Q frags resident), online softmax, publish P+alpha.
// Both waves: O += P*V on their half (8 c-tiles each).
// K tile staged via async global->LDS when available.
// r_ws output layout: [b][c][n] f32.
// ---------------------------------------------------------------------------
__global__ __launch_bounds__(512) void sagan_attn(const __bf16* __restrict__ Qb,
                                                  const __bf16* __restrict__ Kb,
                                                  const __bf16* __restrict__ Vb,
                                                  float* __restrict__ r_ws) {
  const int bi = blockIdx.x;
  const int q0 = blockIdx.y * 128;
  const int tid = threadIdx.x;
  const int lane = tid & 31, wave = tid >> 5;
  const int pair = wave >> 1;
  const bool producer = (wave & 1) == 0;
  const int chalf = wave & 1;           // which 128-channel half this wave owns
  const int half = (lane >> 4) & 1;
  const int col = lane & 15;
  const int qbase = q0 + pair * 16;

  __shared__ __bf16 Kt[32][264];     // [key][channel]
  __shared__ __bf16 Vt[256][40];     // [channel][key] (transposed stage)
  __shared__ __bf16 Pt[8][16][40];   // per-pair softmax tile [qrow][key]
  __shared__ float  Al[8][16];       // per-pair alpha per query row
  __shared__ float  Ll[8][16];       // per-pair final row sums

  // producer-only resident Q fragments: 16 rows x 256 channels
  v16bf qfr[8];
  if (producer) {
    const __bf16* qrow = Qb + ((size_t)bi * N_ + qbase + col) * C_;
    for (int kk = 0; kk < 8; ++kk) qfr[kk] = load16(qrow + kk * 32 + half * 16);
  }

  v8f O[8];
  for (int ci = 0; ci < 8; ++ci)
    for (int r = 0; r < 8; ++r) O[ci][r] = 0.f;
  float mrow[8], lrow[8];
  for (int r = 0; r < 8; ++r) { mrow[r] = -1e30f; lrow[r] = 0.f; }

  const float scale = 0.0625f;  // 1/sqrt(256)

  for (int kt = 0; kt < N_; kt += 32) {
    // ---- stage K (async->LDS if available) and V (transposed, via VGPRs) ----
    for (int idx = tid; idx < 32 * 32; idx += 512) {
      int key = idx >> 5;
      int ch = (idx & 31) * 8;
      size_t g = ((size_t)bi * N_ + kt + key) * C_ + ch;
#ifdef HAVE_ASYNC_LDS
      __builtin_amdgcn_global_load_async_to_lds_b128(
          (gv4i*)(Kb + g), (lv4i*)&Kt[key][ch], 0, 0);
#else
      *(uint4*)&Kt[key][ch] = *(const uint4*)&Kb[g];
#endif
      union { uint4 u; __bf16 h[8]; } vv;
      vv.u = *(const uint4*)&Vb[g];
      for (int j = 0; j < 8; ++j) Vt[ch + j][key] = vv.h[j];
    }
#ifdef HAVE_ASYNC_LDS
    wait_asynccnt0();  // ASYNCcnt is per-wave; must drain before the barrier
#endif
    if (kt + 32 < N_)
      __builtin_prefetch(&Kb[((size_t)bi * N_ + kt + 32) * C_] + (size_t)tid * 16, 0, 1);
    __syncthreads();

    // ---- producer: S = Q*K^T, online softmax, publish P and alpha ----
    float al[8];
    if (producer) {
      v8f S0, S1;
      for (int r = 0; r < 8; ++r) { S0[r] = 0.f; S1[r] = 0.f; }
      for (int kk = 0; kk < 8; ++kk) {
        v16bf b0 = load16(&Kt[col][kk * 32 + half * 16]);
        v16bf b1 = load16(&Kt[16 + col][kk * 32 + half * 16]);
        S0 = wmma_bf16(qfr[kk], b0, S0);
        S1 = wmma_bf16(qfr[kk], b1, S1);
      }
      for (int r = 0; r < 8; ++r) {
        float s0 = S0[r] * scale, s1 = S1[r] * scale;
        float t = fmaxf(s0, s1);
        for (int m = 1; m <= 8; m <<= 1) t = fmaxf(t, __shfl_xor(t, m, 32));
        float mn = fmaxf(mrow[r], t);
        al[r] = __expf(mrow[r] - mn);
        mrow[r] = mn;
        float p0 = __expf(s0 - mn), p1 = __expf(s1 - mn);
        float s = p0 + p1;
        for (int m = 1; m <= 8; m <<= 1) s += __shfl_xor(s, m, 32);
        lrow[r] = lrow[r] * al[r] + s;
        Pt[pair][r + 8 * half][col] = (__bf16)p0;
        Pt[pair][r + 8 * half][16 + col] = (__bf16)p1;
      }
      if (col == 0)
        for (int r = 0; r < 8; ++r) Al[pair][r + 8 * half] = al[r];
    }
    __syncthreads();
    if (!producer)
      for (int r = 0; r < 8; ++r) al[r] = Al[pair][r + 8 * half];

    // ---- both waves: O = O*alpha + P*V on their 8 c-tiles ----
    v16bf pa = load16(&Pt[pair][col][half * 16]);
    for (int ci = 0; ci < 8; ++ci) {
      v8f cacc;
      for (int r = 0; r < 8; ++r) cacc[r] = O[ci][r] * al[r];
      v16bf vb = load16(&Vt[(chalf * 8 + ci) * 16 + col][half * 16]);
      O[ci] = wmma_bf16(pa, vb, cacc);
    }
    __syncthreads();
  }

  // publish row sums, normalize, store (q is the fast axis of [b][c][n])
  if (producer && col == 0)
    for (int r = 0; r < 8; ++r) Ll[pair][r + 8 * half] = lrow[r];
  __syncthreads();
  float il[8];
  for (int r = 0; r < 8; ++r) il[r] = 1.0f / Ll[pair][r + 8 * half];
  for (int ci = 0; ci < 8; ++ci) {
    int ch = (chalf * 8 + ci) * 16 + col;
    float* base = &r_ws[((size_t)bi * C_ + ch) * N_ + qbase + 8 * half];
    float4 lo, hi;
    lo.x = O[ci][0] * il[0]; lo.y = O[ci][1] * il[1];
    lo.z = O[ci][2] * il[2]; lo.w = O[ci][3] * il[3];
    hi.x = O[ci][4] * il[4]; hi.y = O[ci][5] * il[5];
    hi.z = O[ci][6] * il[6]; hi.w = O[ci][7] * il[7];
    *(float4*)(base) = lo;
    *(float4*)(base + 4) = hi;
  }
}

// ---------------------------------------------------------------------------
// Final BN+ReLU+W_o GEMM + residual, f32 output [b][c][h*w]
// ---------------------------------------------------------------------------
__global__ __launch_bounds__(256) void sagan_out_proj(
    const float* __restrict__ r_ws, const float* __restrict__ x,
    const __bf16* __restrict__ Wob, const float* __restrict__ go,
    const float* __restrict__ bo, const float* __restrict__ mean,
    const float* __restrict__ rstd, float* __restrict__ out) {
  const int bi = blockIdx.x;
  const int t0 = blockIdx.y * 64;
  const int tid = threadIdx.x;
  const int lane = tid & 31, wave = tid >> 5;
  const int half = (lane >> 4) & 1;
  const int col = lane & 15;

  __shared__ __bf16 HT[64][264];

  for (int idx = tid; idx < 64 * 256 / 4; idx += 256) {
    int tok4 = idx & 15;
    int ch = idx >> 4;
    float4 rv = *(const float4*)&r_ws[((size_t)bi * C_ + ch) * N_ + t0 + tok4 * 4];
    float gg = go[ch], bb = bo[ch], mm = mean[ch], rr = rstd[ch];
    HT[tok4 * 4 + 0][ch] = (__bf16)fmaxf((rv.x - mm) * rr * gg + bb, 0.f);
    HT[tok4 * 4 + 1][ch] = (__bf16)fmaxf((rv.y - mm) * rr * gg + bb, 0.f);
    HT[tok4 * 4 + 2][ch] = (__bf16)fmaxf((rv.z - mm) * rr * gg + bb, 0.f);
    HT[tok4 * 4 + 3][ch] = (__bf16)fmaxf((rv.w - mm) * rr * gg + bb, 0.f);
  }
  __syncthreads();

  for (int oi = wave * 2; oi < wave * 2 + 2; ++oi) {
    v16bf afr[8];
    const __bf16* Wrow = Wob + (size_t)(oi * 16 + col) * C_;
    for (int kk = 0; kk < 8; ++kk) afr[kk] = load16(Wrow + kk * 32 + half * 16);
    for (int tt = 0; tt < 4; ++tt) {
      v8f acc;
      for (int i = 0; i < 8; ++i) acc[i] = 0.f;
      for (int kk = 0; kk < 8; ++kk) {
        v16bf bf = load16(&HT[tt * 16 + col][kk * 32 + half * 16]);
        acc = wmma_bf16(afr[kk], bf, acc);
      }
      int tok = t0 + tt * 16 + col;
      for (int r = 0; r < 8; ++r) {
        int orow = oi * 16 + r + 8 * half;
        size_t g = ((size_t)bi * C_ + orow) * N_ + tok;
        out[g] = x[g] + acc[r];  // residual (coalesced across lanes via tok)
      }
    }
  }
}

// ---------------------------------------------------------------------------
extern "C" void kernel_launch(void* const* d_in, const int* in_sizes, int n_in,
                              void* d_out, int out_size, void* d_ws, size_t ws_size,
                              hipStream_t stream) {
  const float* x  = (const float*)d_in[0];
  const float* wq = (const float*)d_in[1];
  const float* wk = (const float*)d_in[2];
  const float* wv = (const float*)d_in[3];
  const float* wo = (const float*)d_in[4];
  const float* gq = (const float*)d_in[5];
  const float* bq = (const float*)d_in[6];
  const float* gk = (const float*)d_in[7];
  const float* bk = (const float*)d_in[8];
  const float* gv = (const float*)d_in[9];
  const float* bv = (const float*)d_in[10];
  const float* go = (const float*)d_in[11];
  const float* bo = (const float*)d_in[12];
  float* out = (float*)d_out;

  char* ws = (char*)d_ws;
  float* mean1 = (float*)(ws + 0);
  float* rstd1 = (float*)(ws + 1024);
  float* mean2 = (float*)(ws + 2048);
  float* rstd2 = (float*)(ws + 3072);
  const size_t wb_bytes = (size_t)C_ * C_ * 2;  // 128 KB each
  __bf16* Wqb = (__bf16*)(ws + 4096);
  __bf16* Wkb = (__bf16*)(ws + 4096 + wb_bytes);
  __bf16* Wvb = (__bf16*)(ws + 4096 + 2 * wb_bytes);
  __bf16* Wob = (__bf16*)(ws + 4096 + 3 * wb_bytes);
  const size_t qkv_off = 4096 + 4 * wb_bytes;
  const size_t qkv_bytes = (size_t)B_ * N_ * C_ * 2;  // 16 MB each
  __bf16* Qb = (__bf16*)(ws + qkv_off);
  __bf16* Kb = (__bf16*)(ws + qkv_off + qkv_bytes);
  __bf16* Vb = (__bf16*)(ws + qkv_off + 2 * qkv_bytes);
  float* r_ws = (float*)(ws + qkv_off + 3 * qkv_bytes);

  sagan_wcvt<<<64, 256, 0, stream>>>(wq, wk, wv, wo, Wqb, Wkb, Wvb, Wob);
  sagan_bn_stats<<<C_, 256, 0, stream>>>(x, mean1, rstd1);
  sagan_qkv_proj<<<dim3(B_, N_ / 64), 256, 0, stream>>>(
      x, Wqb, Wkb, Wvb, gq, bq, gk, bk, gv, bv, mean1, rstd1, Qb, Kb, Vb);
  sagan_attn<<<dim3(B_, N_ / 128), 512, 0, stream>>>(Qb, Kb, Vb, r_ws);
  sagan_bn_stats<<<C_, 256, 0, stream>>>(r_ws, mean2, rstd2);
  sagan_out_proj<<<dim3(B_, N_ / 64), 256, 0, stream>>>(
      r_ws, x, Wob, go, bo, mean2, rstd2, out);
}
